// BackboneFeatureExtractor_1906965479586
// MI455X (gfx1250) — matrix operand order repaired and checked by reading, not compiled, not run on hardware
//
#include <hip/hip_runtime.h>
#include <math.h>

// ---------------------------------------------------------------------------
// MI455X / gfx1250 (CDNA5, wave32) implementation.
// Heavy GEMMs + attention on v_wmma_f32_16x16x32_f16; GEMM tiles staged with
// gfx1250 async global->LDS copies (ASYNCcnt) and double-buffered LDS.
// ---------------------------------------------------------------------------

typedef __attribute__((ext_vector_type(16))) _Float16 v16h;
typedef __attribute__((ext_vector_type(8)))  float    v8f;

#define WMMA_F16(A, B, C) \
  __builtin_amdgcn_wmma_f32_16x16x32_f16(false, (A), false, (B), (short)0, (C), false, false)

// 16B global -> LDS async copy (per-lane), tracked by ASYNCcnt.
__device__ __forceinline__ void async_copy16(const _Float16* g, _Float16* l) {
  unsigned loff = (unsigned)(unsigned long long)(void*)l;  // flat addr low 32b == LDS offset
  asm volatile("global_load_async_to_lds_b128 %0, %1, off"
               :: "v"(loff), "v"(g) : "memory");
}
__device__ __forceinline__ void async_wait0() {
  asm volatile("s_wait_asynccnt 0x0" ::: "memory");
}

// ---------------- small prep kernels ----------------------------------------

// out_f16[n,k] = W[n,k] + s * sum_r B[n,r] * A[r,k]   (fold rank-4 LoRA into W)
__global__ void fuse_lora_f16(const float* __restrict__ W, const float* __restrict__ A,
                              const float* __restrict__ Bm, _Float16* __restrict__ out,
                              int N, int K, float s) {
  int i = blockIdx.x * 256 + threadIdx.x;
  if (i >= N * K) return;
  int n = i / K, k = i % K;
  float acc = W[i];
#pragma unroll
  for (int r = 0; r < 4; ++r) acc += s * Bm[n * 4 + r] * A[r * K + k];
  out[i] = (_Float16)acc;
}

__global__ void cvt_f16(const float* __restrict__ x, _Float16* __restrict__ y, int n) {
  int i = blockIdx.x * 256 + threadIdx.x;
  if (i < n) y[i] = (_Float16)x[i];
}

// encoder (4,77,768) -> padded (4*80, 768) f16, zero pad rows 77..79 per batch
__global__ void pad_enc_f16(const float* __restrict__ enc, _Float16* __restrict__ out) {
  int i = blockIdx.x * 256 + threadIdx.x;
  if (i >= 320 * 768) return;
  int row = i / 768, k = i % 768;
  int b = row / 80, t = row % 80;
  out[i] = (t < 77) ? (_Float16)enc[(b * 77 + t) * 768 + k] : (_Float16)0.0f;
}

// ---------------- WMMA GEMM: C(M,N) = X(M,K) @ W(N,K)^T ----------------------
// f16 in, f32 accum; async double-buffered LDS staging; compile-time epilogue.
// M % BM == 0, N % BN == 0, K % 32 == 0.
template <int BM, int BN, int WM_, int WN_, int WMF, int WNF,
          bool GELU, bool HAS_BIAS, bool HAS_RES, bool STORE_F, bool STORE_H>
__global__ __launch_bounds__(256) void gemm_wmma(
    const _Float16* __restrict__ X, const _Float16* __restrict__ W,
    const float* __restrict__ bias, const float* __restrict__ res,
    float* __restrict__ outF, _Float16* __restrict__ outH, int M, int N, int K) {
  static_assert(WM_ * WN_ == 8, "8 waves");
  static_assert(BM == WM_ * WMF * 16 && BN == WN_ * WNF * 16, "tiling");
  __shared__ __align__(16) _Float16 sA[2][BM][40];  // 32 valid cols, stride 20 banks
  __shared__ __align__(16) _Float16 sB[2][BN][40];

  const int tid = threadIdx.x;
  const int lane = tid & 31, wave = tid >> 5;
  const int wm = wave % WM_, wn = wave / WM_;
  const int m0 = blockIdx.x * BM, n0 = blockIdx.y * BN;

  v8f zero = {};
  v8f acc[WMF][WNF];
#pragma unroll
  for (int i = 0; i < WMF; ++i)
#pragma unroll
    for (int j = 0; j < WNF; ++j) acc[i][j] = zero;

  const int aoff = (lane < 16) ? 0 : 8;   // A-matrix K chunk base per half-wave
  const int boff = (lane < 16) ? 0 : 16;  // unused: B frags also use aoff as K base
  const int KT = K >> 5;

  // async stage of one 32-wide K tile into LDS buffer `buf`
  auto stage = [&](int kt, int buf) {
#pragma unroll
    for (int it = 0; it < BM / 64; ++it) {
      int idx = it * 2048 + tid * 8;
      int r = idx >> 5, c = idx & 31;
      async_copy16(&X[(size_t)(m0 + r) * K + kt * 32 + c], &sA[buf][r][c]);
    }
#pragma unroll
    for (int it = 0; it < BN / 64; ++it) {
      int idx = it * 2048 + tid * 8;
      int r = idx >> 5, c = idx & 31;
      async_copy16(&W[(size_t)(n0 + r) * K + kt * 32 + c], &sB[buf][r][c]);
    }
  };

  stage(0, 0);
  async_wait0();
  __syncthreads();

  for (int kt = 0; kt < KT; ++kt) {
    const int cur = kt & 1;
    if (kt + 1 < KT) stage(kt + 1, cur ^ 1);  // prefetch next tile (other buffer)

    v16h af[WMF], bf[WNF];
#pragma unroll
    for (int i = 0; i < WMF; ++i) {
      const _Float16* p = &sA[cur][wm * WMF * 16 + i * 16 + (lane & 15)][0];
      union { v16h h; uint4 u[2]; } f;
      f.u[0] = *(const uint4*)(p + aoff);       // K = aoff..aoff+7
      f.u[1] = *(const uint4*)(p + aoff + 16);  // K = aoff+16..aoff+23
      af[i] = f.h;
    }
#pragma unroll
    for (int j = 0; j < WNF; ++j) {
      const _Float16* p = &sB[cur][wn * WNF * 16 + j * 16 + (lane & 15)][0];
      union { v16h h; uint4 u[2]; } f;
      f.u[0] = *(const uint4*)(p + aoff);       // K = aoff..aoff+7
      f.u[1] = *(const uint4*)(p + aoff + 16);
      bf[j] = f.h;
    }
#pragma unroll
    for (int i = 0; i < WMF; ++i)
#pragma unroll
      for (int j = 0; j < WNF; ++j) acc[i][j] = WMMA_F16(af[i], bf[j], acc[i][j]);

    async_wait0();    // our prefetch for kt+1 has landed
    __syncthreads();  // everyone done reading `cur`, prefetches visible
  }

  // epilogue: C layout row = v + 8*(lane>=16), col = lane%16
  const int rhalf = (lane < 16) ? 0 : 8;
#pragma unroll
  for (int i = 0; i < WMF; ++i) {
    const int rbase = m0 + wm * WMF * 16 + i * 16 + rhalf;
#pragma unroll
    for (int j = 0; j < WNF; ++j) {
      const int col = n0 + wn * WNF * 16 + j * 16 + (lane & 15);
      const float bv = HAS_BIAS ? bias[col] : 0.0f;
      v8f a = acc[i][j];
#pragma unroll
      for (int v = 0; v < 8; ++v) {
        size_t o = (size_t)(rbase + v) * N + col;
        float x = a[v] + bv;
        if (GELU) x = 0.5f * x * (1.0f + erff(x * 0.70710678118654752f));
        if (HAS_RES) x += res[o];
        if (STORE_F) outF[o] = x;
        if (STORE_H) outH[o] = (_Float16)x;
      }
    }
  }
}

// ---------------- WMMA attention --------------------------------------------
// One block per (64-query chunk, b,h): 4 waves, each owns one 16-query tile.
// K staged row-major (keys x 160), V staged transposed (160 x keys) in LDS.
// TPAD = padded key count (96 for global T=77, 32 for entity lists).
template <int TPAD, bool STORE_F>
__global__ __launch_bounds__(128) void attn_wmma(
    const _Float16* __restrict__ Qh,  // (B*4096, 1280)
    const _Float16* __restrict__ Kh,  // (B*80,   1280) padded
    const _Float16* __restrict__ Vh,  // (B*80,   1280) padded
    const int* __restrict__ toks, int ntoks, float scale,
    float* __restrict__ outF, _Float16* __restrict__ outH) {
  constexpr int D = 1280, HD = 160, S = 4096;
  constexpr int NT = TPAD / 16;  // score column tiles
  constexpr int KS = TPAD / 32;  // output K steps
  const int tid = threadIdx.x, lane = tid & 31, wave = tid >> 5;
  const int bh = blockIdx.y, b = bh >> 3, h = bh & 7;

  __shared__ __align__(16) _Float16 sK[TPAD][HD + 8];     // stride 84 banks
  __shared__ __align__(16) _Float16 sVt[HD][TPAD + 8];    // stride 52 banks
  __shared__ __align__(16) _Float16 sP[4][16][TPAD + 8];  // per-wave probs

  // stage K (row-major) and V (transposed); zero-fill padded keys
  for (int e = tid * 8; e < TPAD * HD; e += 128 * 8) {
    int j = e / HD, c = e % HD;
    uint4 kq = {0u, 0u, 0u, 0u}, vq = {0u, 0u, 0u, 0u};
    if (j < ntoks) {
      int krow = b * 80 + (toks ? toks[j] : j);
      kq = *(const uint4*)&Kh[(size_t)krow * D + h * HD + c];
      vq = *(const uint4*)&Vh[(size_t)krow * D + h * HD + c];
    }
    *(uint4*)&sK[j][c] = kq;
    const _Float16* vh = (const _Float16*)&vq;
#pragma unroll
    for (int u = 0; u < 8; ++u) sVt[c + u][j] = vh[u];
  }
  __syncthreads();

  const int m0 = (blockIdx.x * 4 + wave) * 16;
  const int aoff = (lane < 16) ? 0 : 8;
  const int boff = (lane < 16) ? 0 : 16;
  const int rhalf = (lane < 16) ? 0 : 8;

  // Q fragments for the whole 160-dim reduction (5 x K=32)
  v16h qa[5];
  const size_t qbase = (size_t)(b * S + m0 + (lane & 15)) * D + h * HD;
#pragma unroll
  for (int kc = 0; kc < 5; ++kc) {
    const _Float16* p = &Qh[qbase + kc * 32];
    union { v16h h; uint4 u[2]; } f;
    f.u[0] = *(const uint4*)(p + aoff);
    f.u[1] = *(const uint4*)(p + aoff + 16);
    qa[kc] = f.h;
  }

  // scores S = Q K^T * scale, masked beyond ntoks
  v8f sc[NT];
#pragma unroll
  for (int nt = 0; nt < NT; ++nt) {
    v8f c = {};
#pragma unroll
    for (int kc = 0; kc < 5; ++kc) {
      const _Float16* p = &sK[nt * 16 + (lane & 15)][kc * 32 + aoff];
      union { v16h h; uint4 u[2]; } f;
      f.u[0] = *(const uint4*)p;
      f.u[1] = *(const uint4*)(p + 16);
      c = WMMA_F16(qa[kc], f.h, c);
    }
    int j = nt * 16 + (lane & 15);
#pragma unroll
    for (int v = 0; v < 8; ++v) c[v] = (j < ntoks) ? c[v] * scale : -1e30f;
    sc[nt] = c;
  }

  // softmax over keys: row m = v + rhalf lives in one 16-lane half-wave
  float pr[NT][8];
#pragma unroll
  for (int v = 0; v < 8; ++v) {
    float m = -1e30f;
#pragma unroll
    for (int nt = 0; nt < NT; ++nt) m = fmaxf(m, sc[nt][v]);
#pragma unroll
    for (int off = 1; off < 16; off <<= 1) m = fmaxf(m, __shfl_xor(m, off));
    float s = 0.0f;
#pragma unroll
    for (int nt = 0; nt < NT; ++nt) {
      float e = __expf(sc[nt][v] - m);
      pr[nt][v] = e;
      s += e;
    }
#pragma unroll
    for (int off = 1; off < 16; off <<= 1) s += __shfl_xor(s, off);
    float inv = 1.0f / s;
#pragma unroll
    for (int nt = 0; nt < NT; ++nt) pr[nt][v] *= inv;
  }

  // stash P in per-wave LDS (C layout -> row-major 16 x TPAD)
#pragma unroll
  for (int nt = 0; nt < NT; ++nt)
#pragma unroll
    for (int v = 0; v < 8; ++v)
      sP[wave][v + rhalf][nt * 16 + (lane & 15)] = (_Float16)pr[nt][v];

  // O = P V  (A-frags from sP, B-frags from transposed V)
#pragma unroll
  for (int nd = 0; nd < 10; ++nd) {
    v8f acc = {};
#pragma unroll
    for (int ks = 0; ks < KS; ++ks) {
      const _Float16* pp = &sP[wave][lane & 15][ks * 32 + aoff];
      union { v16h h; uint4 u[2]; } fa;
      fa.u[0] = *(const uint4*)pp;
      fa.u[1] = *(const uint4*)(pp + 16);
      const _Float16* pv = &sVt[nd * 16 + (lane & 15)][ks * 32 + aoff];
      union { v16h h; uint4 u[2]; } fb;
      fb.u[0] = *(const uint4*)pv;
      fb.u[1] = *(const uint4*)(pv + 16);
      acc = WMMA_F16(fa.h, fb.h, acc);
    }
#pragma unroll
    for (int v = 0; v < 8; ++v) {
      int row = m0 + v + rhalf;
      size_t o = (size_t)(b * S + row) * D + h * HD + nd * 16 + (lane & 15);
      if (STORE_F) outF[o] = acc[v];
      outH[o] = (_Float16)acc[v];
    }
  }
}

// ---------------- host launch ------------------------------------------------

extern "C" void kernel_launch(void* const* d_in, const int* in_sizes, int n_in,
                              void* d_out, int out_size, void* d_ws, size_t ws_size,
                              hipStream_t stream) {
  const float* hidden = (const float*)d_in[0];   // (4,4096,1280)
  const float* enc    = (const float*)d_in[1];   // (4,77,768)
  const int*   toks0  = (const int*)d_in[2];     // (8,)
  const int*   toks1  = (const int*)d_in[3];     // (12,)
  const float* Wq  = (const float*)d_in[4];
  const float* Wk  = (const float*)d_in[5];
  const float* Wv  = (const float*)d_in[6];
  const float* Wo  = (const float*)d_in[7];
  const float* bo  = (const float*)d_in[8];
  const float* lkA = (const float*)d_in[9];
  const float* lkB = (const float*)d_in[10];
  const float* lvA = (const float*)d_in[11];
  const float* lvB = (const float*)d_in[12];
  const float* loA = (const float*)d_in[13];
  const float* loB = (const float*)d_in[14];
  const float* a0dW = (const float*)d_in[15];
  const float* a0db = (const float*)d_in[16];
  const float* a0uW = (const float*)d_in[17];
  const float* a0ub = (const float*)d_in[18];
  const float* a1dW = (const float*)d_in[19];
  const float* a1db = (const float*)d_in[20];
  const float* a1uW = (const float*)d_in[21];
  const float* a1ub = (const float*)d_in[22];

  const int B = 4, S = 4096, D = 1280, Dt = 768;
  const long MS = (long)B * S;               // 16384
  const float lora_s = 0.25f;                // 1/rank
  const float att_scale = 0.07905694150420949f;  // 160^-0.5

  float* out = (float*)d_out;
  const size_t F0OFF = (size_t)MS * D;       // out | F0 | F1
  const size_t F1OFF = 2 * F0OFF;

  // carve f16 workspace (~228 MB total)
  char* w = (char*)d_ws;
  auto alloc_h = [&](size_t n) {
    _Float16* p = (_Float16*)w;
    w += ((n * sizeof(_Float16) + 255) & ~(size_t)255);
    return p;
  };
  _Float16* wk_h   = alloc_h((size_t)D * Dt);
  _Float16* wv_h   = alloc_h((size_t)D * Dt);
  _Float16* wo_h   = alloc_h((size_t)D * D);
  _Float16* wq_h   = alloc_h((size_t)D * D);
  _Float16* a0d_h  = alloc_h((size_t)64 * D);
  _Float16* a0u_h  = alloc_h((size_t)D * 64);
  _Float16* a1d_h  = alloc_h((size_t)64 * D);
  _Float16* a1u_h  = alloc_h((size_t)D * 64);
  _Float16* hid_h  = alloc_h((size_t)MS * D);
  _Float16* encp_h = alloc_h((size_t)320 * Dt);
  _Float16* q_h    = alloc_h((size_t)MS * D);
  _Float16* k_h    = alloc_h((size_t)320 * D);
  _Float16* v_h    = alloc_h((size_t)320 * D);
  _Float16* fg_h   = alloc_h((size_t)MS * D);
  _Float16* f0_h   = alloc_h((size_t)MS * D);
  _Float16* f1_h   = alloc_h((size_t)MS * D);
  _Float16* h0_h   = alloc_h((size_t)MS * 64);
  _Float16* h1_h   = alloc_h((size_t)MS * 64);

  auto nb = [](long n) { return (unsigned)((n + 255) / 256); };

  // 1) fold LoRA into weights, convert all operands to f16
  fuse_lora_f16<<<nb((long)D * Dt), 256, 0, stream>>>(Wk, lkA, lkB, wk_h, D, Dt, lora_s);
  fuse_lora_f16<<<nb((long)D * Dt), 256, 0, stream>>>(Wv, lvA, lvB, wv_h, D, Dt, lora_s);
  fuse_lora_f16<<<nb((long)D * D), 256, 0, stream>>>(Wo, loA, loB, wo_h, D, D, lora_s);
  cvt_f16<<<nb((long)D * D), 256, 0, stream>>>(Wq, wq_h, D * D);
  cvt_f16<<<nb(64L * D), 256, 0, stream>>>(a0dW, a0d_h, 64 * D);
  cvt_f16<<<nb((long)D * 64), 256, 0, stream>>>(a0uW, a0u_h, D * 64);
  cvt_f16<<<nb(64L * D), 256, 0, stream>>>(a1dW, a1d_h, 64 * D);
  cvt_f16<<<nb((long)D * 64), 256, 0, stream>>>(a1uW, a1u_h, D * 64);
  cvt_f16<<<nb(MS * D), 256, 0, stream>>>(hidden, hid_h, (int)(MS * D));
  pad_enc_f16<<<nb(320L * Dt), 256, 0, stream>>>(enc, encp_h);

  // 2) projections: K, V (M=320 padded), Q (M=16384)
  gemm_wmma<64, 128, 2, 4, 2, 2, false, false, false, false, true>
      <<<dim3(5, 10), 256, 0, stream>>>(
      encp_h, wk_h, nullptr, nullptr, nullptr, k_h, 320, D, Dt);
  gemm_wmma<64, 128, 2, 4, 2, 2, false, false, false, false, true>
      <<<dim3(5, 10), 256, 0, stream>>>(
      encp_h, wv_h, nullptr, nullptr, nullptr, v_h, 320, D, Dt);
  gemm_wmma<128, 128, 4, 2, 2, 4, false, false, false, false, true>
      <<<dim3(128, 10), 256, 0, stream>>>(
      hid_h, wq_h, nullptr, nullptr, nullptr, q_h, (int)MS, D, D);

  // 3) attention: global + two entity-gathered variants
  attn_wmma<96, false><<<dim3(64, 32), 128, 0, stream>>>(
      q_h, k_h, v_h, nullptr, 77, att_scale, nullptr, fg_h);
  attn_wmma<32, true><<<dim3(64, 32), 128, 0, stream>>>(
      q_h, k_h, v_h, toks0, 8, att_scale, out + F0OFF, f0_h);
  attn_wmma<32, true><<<dim3(64, 32), 128, 0, stream>>>(
      q_h, k_h, v_h, toks1, 12, att_scale, out + F1OFF, f1_h);

  // 4) adapters: F = F_raw + up(gelu(down(F_raw)))   (residual in-place in d_out)
  gemm_wmma<256, 64, 8, 1, 2, 4, true, true, false, false, true>
      <<<dim3(64, 1), 256, 0, stream>>>(
      f0_h, a0d_h, a0db, nullptr, nullptr, h0_h, (int)MS, 64, D);
  gemm_wmma<128, 128, 4, 2, 2, 4, false, true, true, true, false>
      <<<dim3(128, 10), 256, 0, stream>>>(
      h0_h, a0u_h, a0ub, out + F0OFF, out + F0OFF, nullptr, (int)MS, D, 64);
  gemm_wmma<256, 64, 8, 1, 2, 4, true, true, false, false, true>
      <<<dim3(64, 1), 256, 0, stream>>>(
      f1_h, a1d_h, a1db, nullptr, nullptr, h1_h, (int)MS, 64, D);
  gemm_wmma<128, 128, 4, 2, 2, 4, false, true, true, true, false>
      <<<dim3(128, 10), 256, 0, stream>>>(
      h1_h, a1u_h, a1ub, out + F1OFF, out + F1OFF, nullptr, (int)MS, D, 64);

  // 5) output projection (LoRA already folded into wo_h) + bias
  gemm_wmma<128, 128, 4, 2, 2, 4, false, true, false, true, false>
      <<<dim3(128, 10), 256, 0, stream>>>(
      fg_h, wo_h, bo, nullptr, out, nullptr, (int)MS, D, D);

  (void)in_sizes; (void)n_in; (void)out_size; (void)ws_size;
}
